// GraphAttentionLayer_40492951667145
// MI455X (gfx1250) — compile-verified, matrix operand order checked by reading, not run on hardware
//
#include <hip/hip_runtime.h>
#include <hip/hip_bf16.h>

#define N_NODES 50000
#define N_EDGES 640000
#define HIDDEN  128
#define HEADS   8
#define HEAD_DIM 16
#define ALPHA   0.2f
#define LN_EPS  1e-5f

typedef _Float16 v16h __attribute__((ext_vector_type(16)));
typedef _Float16 v8h  __attribute__((ext_vector_type(8)));
typedef float    v8f  __attribute__((ext_vector_type(8)));

#define LDSP 136   // LDS pitch in f16 elements (multiple of 8 -> 16B aligned rows, staggers banks)

// ---- ordered-uint encoding so unsigned atomicMax == float max (init 0 == -inf) ----
__device__ __forceinline__ unsigned encOrd(float f) {
    unsigned u = __float_as_uint(f);
    return (u & 0x80000000u) ? ~u : (u | 0x80000000u);
}
__device__ __forceinline__ float decOrd(unsigned u) {
    return (u & 0x80000000u) ? __uint_as_float(u & 0x7FFFFFFFu) : __uint_as_float(~u);
}

// =====================================================================
// K0: pack weights to f16. Wflat[c][k] (c = h*16+j, k = input dim) for the
// node projection; Wcomb[c][k] = [eg_w rows | eb_w rows | zero pad] (144x128).
// Both stored N-major so B fragments load contiguously.
// =====================================================================
__global__ void pack_weights_kernel(const float* __restrict__ W,
                                    const float* __restrict__ eg_w,
                                    const float* __restrict__ eb_w,
                                    _Float16* __restrict__ Wflat,
                                    _Float16* __restrict__ Wcomb) {
    int gid = blockIdx.x * 256 + threadIdx.x;
    if (gid < 128 * 128) {
        int c = gid >> 7, i = gid & 127;
        int h = c >> 4, j = c & 15;
        Wflat[gid] = (_Float16)W[h * 2048 + i * 16 + j];   // W[h][i][j]
    }
    if (gid < 144 * 128) {
        int c = gid >> 7, k = gid & 127;
        float v = 0.f;
        if (c < 128)      v = eg_w[c * 128 + k];
        else if (c < 136) v = eb_w[(c - 128) * 128 + k];
        Wcomb[gid] = (_Float16)v;
    }
}

// =====================================================================
// K1: x_heads = x @ Wflat  (M=50000, N=128, K=128) via v_wmma_f32_16x16x32_f16
// Block: 128 threads = 4 waves, 64 rows per block; wave w -> rows w*16..w*16+15.
// =====================================================================
__global__ void __launch_bounds__(128)
node_proj_wmma(const float* __restrict__ x, const _Float16* __restrict__ Wflat,
               float* __restrict__ x_heads) {
    __shared__ __align__(16) _Float16 ldsA[64 * LDSP];
    __shared__ __align__(16) _Float16 ldsB[128 * LDSP];
    const int tid  = threadIdx.x;
    const int base = blockIdx.x * 64;

    // stage A tile (64x128 f32 -> f16)
    for (int c = tid; c < 64 * 32; c += 128) {
        int r = c >> 5, f4 = c & 31;
        int node = base + r;
        float4 v = make_float4(0.f, 0.f, 0.f, 0.f);
        if (node < N_NODES) v = ((const float4*)x)[node * 32 + f4];
        _Float16* d = &ldsA[r * LDSP + f4 * 4];
        d[0] = (_Float16)v.x; d[1] = (_Float16)v.y;
        d[2] = (_Float16)v.z; d[3] = (_Float16)v.w;
    }
    // stage B (128 N-rows x 128 K, f16, N-major)
    for (int c = tid; c < 128 * 16; c += 128) {
        int n = c >> 4, k8 = c & 15;
        *(v8h*)&ldsB[n * LDSP + k8 * 8] = *(const v8h*)&Wflat[n * 128 + k8 * 8];
    }
    __syncthreads();

    const int wave = tid >> 5, lane = tid & 31;
    const int half = lane >> 4, l16 = lane & 15;
    const int mrow = wave * 16;

    v8f acc[8];
#pragma unroll
    for (int t = 0; t < 8; t++) { v8f z = {}; acc[t] = z; }

    for (int kk = 0; kk < 128; kk += 32) {
        const _Float16* ap = &ldsA[(mrow + l16) * LDSP + kk + half * 8];
        v8h alo = *(const v8h*)ap;
        v8h ahi = *(const v8h*)(ap + 16);
        v16h af;
#pragma unroll
        for (int i = 0; i < 8; i++) { af[i] = alo[i]; af[i + 8] = ahi[i]; }
#pragma unroll
        for (int t = 0; t < 8; t++) {
            const _Float16* bp = &ldsB[(t * 16 + l16) * LDSP + kk + half * 8];
            v8h blo = *(const v8h*)bp;
            v8h bhi = *(const v8h*)(bp + 16);
            v16h bf;
#pragma unroll
            for (int i = 0; i < 8; i++) { bf[i] = blo[i]; bf[i + 8] = bhi[i]; }
            acc[t] = __builtin_amdgcn_wmma_f32_16x16x32_f16(
                false, af, false, bf, (short)0, acc[t], false, false);
        }
    }
#pragma unroll
    for (int t = 0; t < 8; t++) {
#pragma unroll
        for (int r = 0; r < 8; r++) {
            int node = base + mrow + r + half * 8;
            int col  = t * 16 + l16;
            if (node < N_NODES) x_heads[node * 128 + col] = acc[t][r];
        }
    }
}

// =====================================================================
// K2: per-node attention half-scores: s_src[n,h]=<x_heads[n,h,:],a_src[h]>,
//     s_dst likewise (math-equivalent to the reference's per-edge einsum).
// =====================================================================
__global__ void __launch_bounds__(256)
node_scores(const float* __restrict__ xh, const float* __restrict__ a,
            float* __restrict__ s_src, float* __restrict__ s_dst) {
    int gid = blockIdx.x * 256 + threadIdx.x;
    if (gid >= N_NODES * HEADS) return;
    int n = gid >> 3, h = gid & 7;
    const float* v  = &xh[n * 128 + h * 16];
    const float* av = &a[h * 32];
    float ss = 0.f, sd = 0.f;
#pragma unroll
    for (int j = 0; j < 16; j++) { ss += v[j] * av[j]; sd += v[j] * av[16 + j]; }
    s_src[gid] = ss; s_dst[gid] = sd;
}

// =====================================================================
// K3: edge WMMA kernel. One pass over edge_attr computes gate logits (128)
// and bias (8) via the fused 128x144 weight matrix. Epilogue: sigmoid->f16
// gate store, e = bias + s_src[row] + s_dst[col] (+eb_b), leaky-ReLU,
// store e, atomic ordered-max per (dst node, head).
// =====================================================================
__global__ void __launch_bounds__(128)
edge_wmma(const float* __restrict__ edge_attr, const _Float16* __restrict__ Wcomb,
          const int* __restrict__ eidx,           // [2, N_EDGES]
          const float* __restrict__ s_src, const float* __restrict__ s_dst,
          const float* __restrict__ eb_b, const float* __restrict__ eg_b,
          _Float16* __restrict__ gate, float* __restrict__ e_buf,
          unsigned* __restrict__ node_max_u) {
    __shared__ __align__(16) _Float16 ldsA[64 * LDSP];
    __shared__ __align__(16) _Float16 ldsB[144 * LDSP];
    const int tid  = threadIdx.x;
    const int base = blockIdx.x * 64;

    for (int c = tid; c < 64 * 32; c += 128) {
        int r = c >> 5, f4 = c & 31;
        int e = base + r;
        float4 v = make_float4(0.f, 0.f, 0.f, 0.f);
        if (e < N_EDGES) v = ((const float4*)edge_attr)[(size_t)e * 32 + f4];
        _Float16* d = &ldsA[r * LDSP + f4 * 4];
        d[0] = (_Float16)v.x; d[1] = (_Float16)v.y;
        d[2] = (_Float16)v.z; d[3] = (_Float16)v.w;
    }
    for (int c = tid; c < 144 * 16; c += 128) {
        int n = c >> 4, k8 = c & 15;
        *(v8h*)&ldsB[n * LDSP + k8 * 8] = *(const v8h*)&Wcomb[n * 128 + k8 * 8];
    }
    __syncthreads();

    const int wave = tid >> 5, lane = tid & 31;
    const int half = lane >> 4, l16 = lane & 15;
    const int mrow = wave * 16;

    v8f acc[9];
#pragma unroll
    for (int t = 0; t < 9; t++) { v8f z = {}; acc[t] = z; }

    for (int kk = 0; kk < 128; kk += 32) {
        const _Float16* ap = &ldsA[(mrow + l16) * LDSP + kk + half * 8];
        v8h alo = *(const v8h*)ap;
        v8h ahi = *(const v8h*)(ap + 16);
        v16h af;
#pragma unroll
        for (int i = 0; i < 8; i++) { af[i] = alo[i]; af[i + 8] = ahi[i]; }
#pragma unroll
        for (int t = 0; t < 9; t++) {
            const _Float16* bp = &ldsB[(t * 16 + l16) * LDSP + kk + half * 8];
            v8h blo = *(const v8h*)bp;
            v8h bhi = *(const v8h*)(bp + 16);
            v16h bf;
#pragma unroll
            for (int i = 0; i < 8; i++) { bf[i] = blo[i]; bf[i + 8] = bhi[i]; }
            acc[t] = __builtin_amdgcn_wmma_f32_16x16x32_f16(
                false, af, false, bf, (short)0, acc[t], false, false);
        }
    }

    const int* erow = eidx;
    const int* ecol = eidx + N_EDGES;

    // gate = sigmoid(logits + eg_b) stored as f16
#pragma unroll
    for (int t = 0; t < 8; t++) {
        int c  = t * 16 + l16;
        float bc = eg_b[c];
#pragma unroll
        for (int r = 0; r < 8; r++) {
            int e = base + mrow + r + half * 8;
            if (e < N_EDGES) {
                float g = 1.f / (1.f + __expf(-(acc[t][r] + bc)));
                gate[(size_t)e * 128 + c] = (_Float16)g;
            }
        }
    }
    // attention logits from tile 8 (cols 128..135 == heads 0..7)
    {
        int h = l16;
        if (h < 8) {
            float bb = eb_b[h];
#pragma unroll
            for (int r = 0; r < 8; r++) {
                int e = base + mrow + r + half * 8;
                if (e < N_EDGES) {
                    int rn = erow[e], cn = ecol[e];
                    float ev = acc[8][r] + bb + s_src[rn * 8 + h] + s_dst[cn * 8 + h];
                    ev = ev > 0.f ? ev : ALPHA * ev;
                    e_buf[e * 8 + h] = ev;
                    atomicMax(&node_max_u[cn * 8 + h], encOrd(ev));
                }
            }
        }
    }
}

// =====================================================================
// K4: exp(e - max) accumulated per (dst node, head)
// =====================================================================
__global__ void __launch_bounds__(256)
softmax_sum(const float* __restrict__ e_buf, const int* __restrict__ ecol,
            const unsigned* __restrict__ nmax, float* __restrict__ nsum) {
    int gid = blockIdx.x * 256 + threadIdx.x;
    if (gid >= N_EDGES * HEADS) return;
    int e = gid >> 3, h = gid & 7;
    int cn = ecol[e];
    float p = __expf(e_buf[gid] - decOrd(nmax[cn * 8 + h]));
    atomicAdd(&nsum[cn * 8 + h], p);
}

// =====================================================================
// K5: msg = src * gate * attn, scatter-add into out_heads (L2-resident)
// =====================================================================
__global__ void __launch_bounds__(256)
scatter_msg(const float* __restrict__ xh, const _Float16* __restrict__ gate,
            const float* __restrict__ e_buf, const int* __restrict__ erow,
            const int* __restrict__ ecol, const unsigned* __restrict__ nmax,
            const float* __restrict__ nsum, float* __restrict__ out_heads) {
    long long gid = (long long)blockIdx.x * 256 + threadIdx.x;
    if (gid >= (long long)N_EDGES * 128) return;
    int e = (int)(gid >> 7), c = (int)(gid & 127), h = c >> 4;
    int rn = erow[e], cn = ecol[e];
    float m = decOrd(nmax[cn * 8 + h]);
    float s = nsum[cn * 8 + h];
    float attn = __expf(e_buf[e * 8 + h] - m) / fmaxf(s, 1e-12f);
    float msg = xh[rn * 128 + c] * (float)gate[(size_t)e * 128 + c] * attn;
    atomicAdd(&out_heads[cn * 128 + c], msg);
}

// =====================================================================
// K6: residual + LayerNorm; one wave32 per node (4 channels per lane)
// =====================================================================
__global__ void __launch_bounds__(256)
ln_out(const float* __restrict__ out_heads, const float* __restrict__ x,
       const float* __restrict__ g, const float* __restrict__ b,
       float* __restrict__ out) {
    int wave = threadIdx.x >> 5, lane = threadIdx.x & 31;
    int n = blockIdx.x * 8 + wave;
    if (n >= N_NODES) return;
    float4 v  = ((const float4*)out_heads)[n * 32 + lane];
    float4 xv = ((const float4*)x)[n * 32 + lane];
    v.x += xv.x; v.y += xv.y; v.z += xv.z; v.w += xv.w;
    float s  = v.x + v.y + v.z + v.w;
    float sq = v.x * v.x + v.y * v.y + v.z * v.z + v.w * v.w;
    for (int m = 16; m > 0; m >>= 1) {
        s  += __shfl_xor(s,  m, 32);
        sq += __shfl_xor(sq, m, 32);
    }
    float mu  = s * (1.f / 128.f);
    float var = sq * (1.f / 128.f) - mu * mu;
    float inv = rsqrtf(var + LN_EPS);
    float4 gv = ((const float4*)g)[lane];
    float4 bv = ((const float4*)b)[lane];
    float4 o;
    o.x = (v.x - mu) * inv * gv.x + bv.x;
    o.y = (v.y - mu) * inv * gv.y + bv.y;
    o.z = (v.z - mu) * inv * gv.z + bv.z;
    o.w = (v.w - mu) * inv * gv.w + bv.w;
    ((float4*)out)[n * 32 + lane] = o;
}

// =====================================================================
extern "C" void kernel_launch(void* const* d_in, const int* in_sizes, int n_in,
                              void* d_out, int out_size, void* d_ws, size_t ws_size,
                              hipStream_t stream) {
    const float* x         = (const float*)d_in[0];
    const int*   edge_idx  = (const int*)d_in[1];
    const float* edge_attr = (const float*)d_in[2];
    const float* W         = (const float*)d_in[3];
    const float* a         = (const float*)d_in[4];
    const float* eb_w      = (const float*)d_in[5];
    const float* eb_b      = (const float*)d_in[6];
    const float* eg_w      = (const float*)d_in[7];
    const float* eg_b      = (const float*)d_in[8];
    const float* ln_g      = (const float*)d_in[9];
    const float* ln_b      = (const float*)d_in[10];
    float* out = (float*)d_out;

    // workspace layout (256B aligned)
    auto al = [](size_t v) { return (v + 255) & ~(size_t)255; };
    char* ws = (char*)d_ws;
    size_t off = 0;
    _Float16* Wflat = (_Float16*)(ws + off); off = al(off + 128 * 128 * 2);
    _Float16* Wcomb = (_Float16*)(ws + off); off = al(off + 144 * 128 * 2);
    float* x_heads  = (float*)(ws + off);    off = al(off + (size_t)N_NODES * 128 * 4);
    float* s_src    = (float*)(ws + off);    off = al(off + (size_t)N_NODES * 8 * 4);
    float* s_dst    = (float*)(ws + off);    off = al(off + (size_t)N_NODES * 8 * 4);
    float* e_buf    = (float*)(ws + off);    off = al(off + (size_t)N_EDGES * 8 * 4);
    _Float16* gate  = (_Float16*)(ws + off); off = al(off + (size_t)N_EDGES * 128 * 2);
    // the next three are contiguous so one memset clears them all
    size_t zoff = off;
    unsigned* node_max_u = (unsigned*)(ws + off); off = al(off + (size_t)N_NODES * 8 * 4);
    float* node_sum      = (float*)(ws + off);    off = al(off + (size_t)N_NODES * 8 * 4);
    float* out_heads     = (float*)(ws + off);    off = al(off + (size_t)N_NODES * 128 * 4);
    size_t zbytes = off - zoff;

    const int* erow = edge_idx;
    const int* ecol = edge_idx + N_EDGES;

    hipMemsetAsync(ws + zoff, 0, zbytes, stream);   // max(enc)=0 == -inf, sums/accum = 0

    pack_weights_kernel<<<(144 * 128 + 255) / 256, 256, 0, stream>>>(W, eg_w, eb_w, Wflat, Wcomb);

    node_proj_wmma<<<(N_NODES + 63) / 64, 128, 0, stream>>>(x, Wflat, x_heads);

    node_scores<<<(N_NODES * 8 + 255) / 256, 256, 0, stream>>>(x_heads, a, s_src, s_dst);

    edge_wmma<<<(N_EDGES + 63) / 64, 128, 0, stream>>>(
        edge_attr, Wcomb, edge_idx, s_src, s_dst, eb_b, eg_b, gate, e_buf, node_max_u);

    softmax_sum<<<(N_EDGES * 8 + 255) / 256, 256, 0, stream>>>(e_buf, ecol, node_max_u, node_sum);

    scatter_msg<<<(int)(((long long)N_EDGES * 128 + 255) / 256), 256, 0, stream>>>(
        x_heads, gate, e_buf, erow, ecol, node_max_u, node_sum, out_heads);

    ln_out<<<(N_NODES + 7) / 8, 256, 0, stream>>>(out_heads, x, ln_g, ln_b, out);
}
